// VGGNet7_binary_32633161515582
// MI455X (gfx1250) — 8-bit matrix operand order repaired and checked by reading, compiled, not run on hardware
//
#include <hip/hip_runtime.h>
#include <hip/hip_bf16.h>
#include <stdint.h>

typedef int v8i __attribute__((ext_vector_type(8)));

#define BN_EPS 1e-5f
#define BATCH 256

__global__ void conv1_direct(const float* __restrict__ x,
                             const float* __restrict__ w,
                             float* __restrict__ out, int total) {
  int idx = blockIdx.x * blockDim.x + threadIdx.x;
  if (idx >= total) return;
  int c  = idx % 128; int t = idx / 128;
  int ow = t % 30;    t /= 30;
  int oh = t % 30;    int b = t / 30;
  float acc = 0.f;
  #pragma unroll
  for (int ci = 0; ci < 3; ++ci)
    #pragma unroll
    for (int kh = 0; kh < 3; ++kh)
      #pragma unroll
      for (int kw = 0; kw < 3; ++kw) {
        float wv = w[((c * 3 + ci) * 3 + kh) * 3 + kw];
        float ws = (wv >= 0.f) ? 1.f : -1.f;
        float xv = x[(((size_t)b * 3 + ci) * 32 + (oh + kh)) * 32 + (ow + kw)];
        acc += ws * xv;
      }
  out[idx] = acc;
}

__global__ void pack_sign_weights(const float* __restrict__ w,
                                  int8_t* __restrict__ pk,
                                  int Cout, int K, int Cin, int KHW,
                                  int Kchunks, int total) {
  int idx = blockIdx.x * blockDim.x + threadIdx.x;
  if (idx >= total) return;
  int j    = idx & 3;
  int i    = (idx >> 2) & 7;
  int lane = (idx >> 5) & 31;
  int blk  = idx >> 10;
  int kc   = blk % Kchunks;
  int mt   = blk / Kchunks;
  int m   = mt * 16 + (lane & 15);
  int kl  = ((i >> 2) << 5) + ((lane >> 4) << 4) + ((i & 3) << 2) + j;
  int kp  = kc * 64 + kl;
  int s   = kp / Cin;
  int cin = kp - s * Cin;
  int8_t v = 0;
  if (m < Cout && kp < K)
    v = (w[(size_t)m * K + (size_t)cin * KHW + s] >= 0.f) ? (int8_t)1
                                                          : (int8_t)-1;
  pk[idx] = v;
}

template <int KH, int KW, int PAD>
__global__ void __launch_bounds__(256) binconv_wmma(
    const int8_t* __restrict__ act,
    const int8_t* __restrict__ pk,
    float* __restrict__ out,
    int Cin, int H, int W, int OH, int OW,
    int CoutReal, int Mtiles, int Ntiles, int Kchunks) {
  constexpr int MT = 4;
  constexpr int NT = 2;
  int lane = threadIdx.x & 31;
  int wv   = threadIdx.x >> 5;
  int mgroups = (Mtiles + MT - 1) / MT;
  int ngroups = Ntiles / NT;
  int tile = blockIdx.x * 8 + wv;
  if (tile >= mgroups * ngroups) return;
  int mg = tile / ngroups;
  int ng = tile % ngroups;

  int OHW = OH * OW;
  int bb[NT], ohh[NT], oww[NT];
  #pragma unroll
  for (int ni = 0; ni < NT; ++ni) {
    int n  = (ng * NT + ni) * 16 + (lane & 15);
    int b  = n / OHW;
    int rp = n - b * OHW;
    int oh = rp / OW;
    bb[ni]  = b;
    ohh[ni] = oh;
    oww[ni] = rp - oh * OW;
  }
  int khalf = (lane >> 4) << 4;
  int cin_chunks = Cin >> 6;

  v8i acc[MT][NT];
  #pragma unroll
  for (int mi = 0; mi < MT; ++mi)
    #pragma unroll
    for (int ni = 0; ni < NT; ++ni)
      acc[mi][ni] = (v8i){0, 0, 0, 0, 0, 0, 0, 0};

  #pragma unroll
  for (int s = 0; s < KH * KW; ++s) {
    int kh = s / KW;
    int kw = s - kh * KW;
    int msk[NT];
    const int8_t* bptr[NT];
    #pragma unroll
    for (int ni = 0; ni < NT; ++ni) {
      int ih = ohh[ni] - PAD + kh;
      int iw = oww[ni] - PAD + kw;
      bool valid = (ih >= 0) && (ih < H) && (iw >= 0) && (iw < W);
      msk[ni] = valid ? ~0 : 0;
      int ihc = ih < 0 ? 0 : (ih >= H ? H - 1 : ih);
      int iwc = iw < 0 ? 0 : (iw >= W ? W - 1 : iw);
      bptr[ni] =
          act + ((((size_t)bb[ni] * H + ihc) * W + iwc) * Cin) + khalf;
    }
    for (int cc = 0; cc < cin_chunks; ++cc) {
      v8i bf[NT];
      #pragma unroll
      for (int ni = 0; ni < NT; ++ni) {
        int4 lo = *(const int4*)(bptr[ni]);
        int4 hi = *(const int4*)(bptr[ni] + 32);
        bptr[ni] += 64;
        bf[ni][0] = lo.x & msk[ni]; bf[ni][1] = lo.y & msk[ni];
        bf[ni][2] = lo.z & msk[ni]; bf[ni][3] = lo.w & msk[ni];
        bf[ni][4] = hi.x & msk[ni]; bf[ni][5] = hi.y & msk[ni];
        bf[ni][6] = hi.z & msk[ni]; bf[ni][7] = hi.w & msk[ni];
      }
      int kc = s * cin_chunks + cc;
      #pragma unroll
      for (int mi = 0; mi < MT; ++mi) {
        int mt = mg * MT + mi;
        if (mt < Mtiles) {
          const v8i* ap = (const v8i*)(pk +
              (((size_t)mt * Kchunks + kc) << 10) + (size_t)lane * 32);
          v8i a = *ap;
          #pragma unroll
          for (int ni = 0; ni < NT; ++ni)
            acc[mi][ni] = __builtin_amdgcn_wmma_i32_16x16x64_iu8(
                true, a, true, bf[ni], acc[mi][ni], false, false);
        }
      }
    }
  }

  int mrow = (lane >> 4) << 3;
  #pragma unroll
  for (int ni = 0; ni < NT; ++ni) {
    size_t pixbase =
        (((size_t)bb[ni] * OH + ohh[ni]) * OW + oww[ni]) * CoutReal;
    #pragma unroll
    for (int mi = 0; mi < MT; ++mi) {
      int mt = mg * MT + mi;
      if (mt < Mtiles) {
        int cbase = mt * 16 + mrow;
        float* op = out + pixbase + cbase;
        if (cbase + 7 < CoutReal) {
          float4 f0 = {(float)acc[mi][ni][0], (float)acc[mi][ni][1],
                       (float)acc[mi][ni][2], (float)acc[mi][ni][3]};
          float4 f1 = {(float)acc[mi][ni][4], (float)acc[mi][ni][5],
                       (float)acc[mi][ni][6], (float)acc[mi][ni][7]};
          *(float4*)(op)     = f0;
          *(float4*)(op + 4) = f1;
        } else {
          #pragma unroll
          for (int r = 0; r < 8; ++r)
            if (cbase + r < CoutReal) op[r] = (float)acc[mi][ni][r];
        }
      }
    }
  }
}

__global__ void zero_buf(float* __restrict__ p, int n) {
  int idx = blockIdx.x * blockDim.x + threadIdx.x;
  if (idx < n) p[idx] = 0.f;
}

__global__ void bn_accum(const float* __restrict__ x, float* __restrict__ acc,
                         int C, long long total) {
  extern __shared__ float sm[];
  for (int i = threadIdx.x; i < 2 * C; i += blockDim.x) sm[i] = 0.f;
  __syncthreads();
  long long stride = (long long)gridDim.x * blockDim.x;
  long long i0 = (long long)blockIdx.x * blockDim.x + threadIdx.x;
  int c    = (int)(i0 % C);
  int step = (int)(stride % C);
  for (long long i = i0; i < total; i += stride) {
    float v = x[i];
    atomicAdd(&sm[c], v);
    atomicAdd(&sm[C + c], v * v);
    c += step;
    if (c >= C) c -= C;
  }
  __syncthreads();
  for (int i = threadIdx.x; i < 2 * C; i += blockDim.x)
    atomicAdd(&acc[i], sm[i]);
}

__global__ void bn_finalize(const float* __restrict__ acc,
                            float* __restrict__ mean,
                            float* __restrict__ invstd, int C, float invN) {
  int c = blockIdx.x * blockDim.x + threadIdx.x;
  if (c >= C) return;
  float m = acc[c] * invN;
  float var = acc[C + c] * invN - m * m;
  mean[c] = m;
  invstd[c] = rsqrtf(var + BN_EPS);
}

__global__ void bn_sign_pack(const float* __restrict__ x,
                             const float* __restrict__ mean,
                             const float* __restrict__ invstd,
                             const float* __restrict__ g,
                             const float* __restrict__ bet,
                             int8_t* __restrict__ actout,
                             int C, long long total) {
  long long stride = (long long)gridDim.x * blockDim.x;
  long long i0 = (long long)blockIdx.x * blockDim.x + threadIdx.x;
  int c    = (int)(i0 % C);
  int step = (int)(stride % C);
  for (long long i = i0; i < total; i += stride) {
    float y = (x[i] - mean[c]) * invstd[c];
    y = y * g[c] + bet[c];
    actout[i] = (y >= 0.f) ? (int8_t)1 : (int8_t)-1;
    c += step;
    if (c >= C) c -= C;
  }
}

__global__ void bn_out_final(const float* __restrict__ x,
                             const float* __restrict__ mean,
                             const float* __restrict__ invstd,
                             float* __restrict__ out, int C, int total) {
  int idx = blockIdx.x * blockDim.x + threadIdx.x;
  if (idx >= total) return;
  int c = idx % C;
  out[idx] = (x[idx] - mean[c]) * invstd[c];
}

__global__ void maxpool2_i8(const int8_t* __restrict__ in,
                            int8_t* __restrict__ out,
                            int C, int IH, int IW, int total) {
  int OW = IW >> 1;
  int OH = IH >> 1;
  int idx = blockIdx.x * blockDim.x + threadIdx.x;
  if (idx >= total) return;
  int c  = idx % C;  int t = idx / C;
  int ow = t % OW;   t /= OW;
  int oh = t % OH;   int b = t / OH;
  size_t row0 = (((size_t)b * IH + 2 * oh) * IW + 2 * ow) * C + c;
  size_t row1 = row0 + (size_t)IW * C;
  int8_t v0 = in[row0], v1 = in[row0 + C];
  int8_t v2 = in[row1], v3 = in[row1 + C];
  int8_t m01 = v0 > v1 ? v0 : v1;
  int8_t m23 = v2 > v3 ? v2 : v3;
  out[idx] = m01 > m23 ? m01 : m23;
}

extern "C" void kernel_launch(void* const* d_in, const int* in_sizes, int n_in,
                              void* d_out, int out_size, void* d_ws,
                              size_t ws_size, hipStream_t stream) {
  const float* x   = (const float*)d_in[0];
  const float* w1  = (const float*)d_in[1];
  const float* wsrc[8] = {(const float*)d_in[2], (const float*)d_in[3],
                          (const float*)d_in[4], (const float*)d_in[5],
                          (const float*)d_in[6], (const float*)d_in[7],
                          (const float*)d_in[8], (const float*)d_in[9]};
  const float* g[8];
  const float* be[8];
  for (int i = 0; i < 8; ++i) {
    g[i]  = (const float*)d_in[10 + 2 * i];
    be[i] = (const float*)d_in[11 + 2 * i];
  }
  float* out = (float*)d_out;

  char* ws = (char*)d_ws;
  size_t off = 0;
  auto carve = [&](size_t bytes) {
    char* p = ws + off;
    off += (bytes + 255) & ~(size_t)255;
    return p;
  };
  float*  convbuf = (float*) carve((size_t)BATCH * 128 * 30 * 30 * 4);
  int8_t* actA    = (int8_t*)carve((size_t)BATCH * 128 * 30 * 30);
  int8_t* actB    = (int8_t*)carve((size_t)BATCH * 128 * 30 * 30);
  float*  mean    = (float*) carve(1024 * 4);
  float*  invstd  = (float*) carve(1024 * 4);
  float*  statacc = (float*) carve(2 * 1024 * 4);

  struct PkInfo { int Cout, Cin, KH, KW, pad; };
  PkInfo pi[8] = {
      {128, 128, 3, 3, 1}, {256, 128, 3, 3, 1}, {256, 256, 3, 3, 1},
      {512, 256, 3, 3, 1}, {512, 512, 3, 3, 1},
      {1024, 512, 3, 3, 0},
      {1024, 1024, 1, 1, 0},
      {10, 1024, 1, 1, 0}};
  int8_t* pk[8];
  for (int l = 0; l < 8; ++l) {
    int KHW     = pi[l].KH * pi[l].KW;
    int K       = pi[l].Cin * KHW;
    int Kchunks = K / 64;
    int Mtiles  = (pi[l].Cout + 15) / 16;
    size_t bytes = (size_t)Mtiles * Kchunks * 1024;
    pk[l] = (int8_t*)carve(bytes);
    int total = (int)bytes;
    pack_sign_weights<<<(total + 255) / 256, 256, 0, stream>>>(
        wsrc[l], pk[l], pi[l].Cout, K, pi[l].Cin, KHW, Kchunks, total);
  }

  auto run_binconv = [&](const int8_t* actin, int l, int H, int W,
                         float* outc) {
    int Cin = pi[l].Cin, KH = pi[l].KH, pad = pi[l].pad;
    int KW  = pi[l].KW;
    int Cout = pi[l].Cout;
    int OH = H + 2 * pad - KH + 1;
    int OW = W + 2 * pad - KW + 1;
    int Kchunks = (Cin * KH * KW) / 64;
    int Mtiles  = (Cout + 15) / 16;
    int Ntiles  = (BATCH * OH * OW) / 16;
    int mgroups = (Mtiles + 3) / 4;
    int ngroups = Ntiles / 2;
    int waves   = mgroups * ngroups;
    int blocks  = (waves + 7) / 8;
    if (KH == 3 && pad == 1)
      binconv_wmma<3, 3, 1><<<blocks, 256, 0, stream>>>(
          actin, pk[l], outc, Cin, H, W, OH, OW, Cout, Mtiles, Ntiles,
          Kchunks);
    else if (KH == 3)
      binconv_wmma<3, 3, 0><<<blocks, 256, 0, stream>>>(
          actin, pk[l], outc, Cin, H, W, OH, OW, Cout, Mtiles, Ntiles,
          Kchunks);
    else
      binconv_wmma<1, 1, 0><<<blocks, 256, 0, stream>>>(
          actin, pk[l], outc, Cin, H, W, OH, OW, Cout, Mtiles, Ntiles,
          Kchunks);
  };
  auto run_stats = [&](const float* cbuf, int C, int HW) {
    zero_buf<<<(2 * C + 255) / 256, 256, 0, stream>>>(statacc, 2 * C);
    long long total = (long long)BATCH * C * HW;
    long long want = (total + 255) / 256;
    int blocks = (int)(want < 2048 ? want : 2048);
    bn_accum<<<blocks, 256, 2 * C * sizeof(float), stream>>>(cbuf, statacc, C,
                                                             total);
    bn_finalize<<<(C + 255) / 256, 256, 0, stream>>>(
        statacc, mean, invstd, C, 1.f / (float)(BATCH * HW));
  };
  auto run_bn_sign = [&](const float* cbuf, int C, int HW, int gi,
                         int8_t* actout) {
    run_stats(cbuf, C, HW);
    long long total = (long long)BATCH * C * HW;
    long long want = (total + 255) / 256;
    int blocks = (int)(want < 4096 ? want : 4096);
    bn_sign_pack<<<blocks, 256, 0, stream>>>(cbuf, mean, invstd, g[gi],
                                             be[gi], actout, C, total);
  };
  auto run_pool = [&](const int8_t* in, int C, int IH, int IW, int8_t* o) {
    int total = BATCH * C * (IH / 2) * (IW / 2);
    maxpool2_i8<<<(total + 255) / 256, 256, 0, stream>>>(in, o, C, IH, IW,
                                                         total);
  };

  {
    int total = BATCH * 128 * 30 * 30;
    conv1_direct<<<(total + 255) / 256, 256, 0, stream>>>(x, w1, convbuf,
                                                          total);
    run_bn_sign(convbuf, 128, 30 * 30, 0, actA);
  }
  run_binconv(actA, 0, 30, 30, convbuf);
  run_bn_sign(convbuf, 128, 30 * 30, 1, actB);
  run_pool(actB, 128, 30, 30, actA);
  run_binconv(actA, 1, 15, 15, convbuf);
  run_bn_sign(convbuf, 256, 15 * 15, 2, actB);
  run_binconv(actB, 2, 15, 15, convbuf);
  run_bn_sign(convbuf, 256, 15 * 15, 3, actA);
  run_pool(actA, 256, 15, 15, actB);
  run_binconv(actB, 3, 7, 7, convbuf);
  run_bn_sign(convbuf, 512, 7 * 7, 4, actA);
  run_binconv(actA, 4, 7, 7, convbuf);
  run_bn_sign(convbuf, 512, 7 * 7, 5, actB);
  run_pool(actB, 512, 7, 7, actA);
  run_binconv(actA, 5, 3, 3, convbuf);
  run_bn_sign(convbuf, 1024, 1, 6, actB);
  run_binconv(actB, 6, 1, 1, convbuf);
  run_bn_sign(convbuf, 1024, 1, 7, actA);
  run_binconv(actA, 7, 1, 1, convbuf);
  run_stats(convbuf, 10, 1);
  {
    int total = BATCH * 10;
    bn_out_final<<<(total + 255) / 256, 256, 0, stream>>>(convbuf, mean,
                                                          invstd, out, 10,
                                                          total);
  }
}
